// BiLSTM_CRF_39316130627946
// MI455X (gfx1250) — compile-verified, matrix operand order checked by reading, not run on hardware
//
#include <hip/hip_runtime.h>
#include <hip/hip_bf16.h>

typedef unsigned short u16;
typedef __attribute__((ext_vector_type(16))) __bf16 v16bf;
typedef __attribute__((ext_vector_type(8)))  __bf16 v8bf;
typedef __attribute__((ext_vector_type(8)))  float  v8f;

#define Bv   64
#define Tv   256
#define Ev   300
#define EP   320
#define Hv   512
#define G4H  2048
#define Lv   21
#define BT   (Bv*Tv)   // 16384

// ---------- helpers ----------
__device__ __forceinline__ u16 f2bf(float f) {
    union { float f; unsigned u; } x; x.f = f;
    unsigned u = x.u;
    unsigned r = (u + 0x7FFFu + ((u >> 16) & 1u)) >> 16;   // RNE
    return (u16)r;
}
__device__ __forceinline__ float bf2f(u16 v) {
    union { unsigned u; float f; } x; x.u = ((unsigned)v) << 16;
    return x.f;
}
__device__ __forceinline__ float sigf(float x) { return 1.0f / (1.0f + __expf(-x)); }

__device__ __forceinline__ v16bf frag_ld(const u16* p0, const u16* p1) {
    v8bf lo = *(const v8bf*)(const void*)p0;
    v8bf hi = *(const v8bf*)(const void*)p1;
    v16bf r;
#pragma unroll
    for (int i = 0; i < 8; ++i) { r[i] = lo[i]; r[i + 8] = hi[i]; }
    return r;
}
__device__ __forceinline__ v8f wmma_bf16(v16bf a, v16bf b, v8f c) {
    return __builtin_amdgcn_wmma_f32_16x16x32_bf16(false, a, false, b, (short)0, c, false, false);
}

// CDNA5 async copy: global memory -> LDS, 16B per lane, tracked by ASYNCcnt.
__device__ __forceinline__ void async_b128(unsigned lds_addr, unsigned long long gaddr) {
    asm volatile("global_load_async_to_lds_b128 %0, %1, off"
                 :: "v"(lds_addr), "v"(gaddr) : "memory");
}
__device__ __forceinline__ void wait_async0() {
    asm volatile("s_wait_asynccnt 0x0" ::: "memory");
}
__device__ __forceinline__ unsigned lds_off(const void* p) {
    return (unsigned)(uintptr_t)p;   // LDS flat address: low 32 bits = LDS offset
}

__device__ __forceinline__ void grid_barrier(int* cnt, int* gen, int nwg, int tid) {
    __syncthreads();
    if (tid == 0) {
        int g = __hip_atomic_load(gen, __ATOMIC_ACQUIRE, __HIP_MEMORY_SCOPE_AGENT);
        int a = __hip_atomic_fetch_add(cnt, 1, __ATOMIC_ACQ_REL, __HIP_MEMORY_SCOPE_AGENT);
        if (a == nwg - 1) {
            __hip_atomic_store(cnt, 0, __ATOMIC_RELAXED, __HIP_MEMORY_SCOPE_AGENT);
            __hip_atomic_fetch_add(gen, 1, __ATOMIC_ACQ_REL, __HIP_MEMORY_SCOPE_AGENT);
        } else {
            while (__hip_atomic_load(gen, __ATOMIC_ACQUIRE, __HIP_MEMORY_SCOPE_AGENT) == g)
                __builtin_amdgcn_s_sleep(1);
        }
    }
    __syncthreads();
}

// ---------- init / conversion ----------
__global__ void zero_init_k(float* out, int* bar) {
    if (blockIdx.x == 0 && threadIdx.x == 0) { out[0] = 0.0f; bar[0] = 0; bar[1] = 0; }
}

__global__ void cvt_pad_k(const float* __restrict__ src, u16* __restrict__ dst,
                          int R, int Ks, int Kp) {
    int idx = blockIdx.x * 256 + threadIdx.x;
    if (idx >= R * Kp) return;
    int r = idx / Kp, k = idx - r * Kp;
    dst[idx] = (k < Ks) ? f2bf(src[(size_t)r * Ks + k]) : (u16)0;
}

__global__ void embed_gather_k(const int* __restrict__ ids,
                               const float* __restrict__ emb,
                               u16* __restrict__ x0) {
    int idx = blockIdx.x * 256 + threadIdx.x;
    if (idx >= BT * EP) return;
    int row = idx / EP, k = idx - row * EP;
    float v = (k < Ev) ? emb[(size_t)ids[row] * Ev + k] : 0.0f;
    x0[idx] = f2bf(v);
}

// ---------- big WMMA GEMM: C[M,N] = A[M,K](bf16) * W[N,K]^T(bf16) + bi + bh ----------
// 128x128 block tile, 64-deep K slices, async double-buffered LDS staging
// (dynamic LDS: 2 x (A 16KB + W 16KB) = 64KB).
#define GM 128
#define GN 128
#define GK 64
__global__ __launch_bounds__(256)
void wmma_gemm_bias_k(const u16* __restrict__ A, int lda,
                      const u16* __restrict__ W, int ldw,
                      const float* __restrict__ bi, const float* __restrict__ bh,
                      float* __restrict__ C, int M, int N, int K) {
    extern __shared__ __align__(16) char gsmem[];
    u16* AsB = (u16*)gsmem;               // 2 x [128][64]
    u16* WsB = (u16*)(gsmem + 32768);     // 2 x [128][64]
    int tid = threadIdx.x;
    int wave = tid >> 5, lane = tid & 31;
    int m0 = blockIdx.y * GM;
    int n0 = blockIdx.x * GN;
    int mtp = wave >> 1;       // m-tile pair: m-tiles {2*mtp, 2*mtp+1}
    int nh  = wave & 1;        // n half (4 n-tiles each)
    v8f acc[2][4] = {};

    // stage 128x64 tiles: each thread moves 4x16B of A and 4x16B of W
    int sr = tid >> 1, sc = (tid & 1) * 32;
    auto stage = [&](int buf, int k0) {
        u16* Ad = AsB + buf * (GM * GK);
        u16* Wd = WsB + buf * (GN * GK);
        const u16* Ag = &A[(size_t)(m0 + sr) * lda + k0 + sc];
        const u16* Wg = &W[(size_t)(n0 + sr) * ldw + k0 + sc];
#pragma unroll
        for (int q = 0; q < 4; ++q) {
            async_b128(lds_off(&Ad[sr * GK + sc + q * 8]),
                       (unsigned long long)(uintptr_t)(Ag + q * 8));
            async_b128(lds_off(&Wd[sr * GK + sc + q * 8]),
                       (unsigned long long)(uintptr_t)(Wg + q * 8));
        }
    };

    int nk = K / GK;
    stage(0, 0);
    wait_async0();
    __syncthreads();
    for (int ki = 0; ki < nk; ++ki) {
        int buf = ki & 1;
        if (ki + 1 < nk) stage(buf ^ 1, (ki + 1) * GK);
        const u16* Ab = AsB + buf * (GM * GK);
        const u16* Wb = WsB + buf * (GN * GK);
        int khalf = lane >> 4, l16 = lane & 15;
#pragma unroll
        for (int kc = 0; kc < GK; kc += 32) {
            v16bf a0 = frag_ld(&Ab[((mtp * 2 + 0) * 16 + l16) * GK + kc + khalf * 8],
                               &Ab[((mtp * 2 + 0) * 16 + l16) * GK + kc + 16 + khalf * 8]);
            v16bf a1 = frag_ld(&Ab[((mtp * 2 + 1) * 16 + l16) * GK + kc + khalf * 8],
                               &Ab[((mtp * 2 + 1) * 16 + l16) * GK + kc + 16 + khalf * 8]);
#pragma unroll
            for (int nt = 0; nt < 4; ++nt) {
                int nr = nh * 64 + nt * 16 + l16;
                v16bf bfm = frag_ld(&Wb[nr * GK + kc + khalf * 8],
                                    &Wb[nr * GK + kc + 16 + khalf * 8]);
                acc[0][nt] = wmma_bf16(a0, bfm, acc[0][nt]);
                acc[1][nt] = wmma_bf16(a1, bfm, acc[1][nt]);
            }
        }
        wait_async0();
        __syncthreads();
    }
    int l16 = lane & 15, hi = lane >> 4;
#pragma unroll
    for (int mi = 0; mi < 2; ++mi) {
#pragma unroll
        for (int nt = 0; nt < 4; ++nt) {
            int col = n0 + nh * 64 + nt * 16 + l16;
            float bias = bi[col] + bh[col];
#pragma unroll
            for (int r = 0; r < 8; ++r) {
                int row = m0 + (mtp * 2 + mi) * 16 + r + 8 * hi;
                C[(size_t)row * N + col] = acc[mi][nt][r] + bias;
            }
        }
    }
}

// ---------- persistent recurrent scan (one direction) ----------
// 32 WGs; WG g owns hidden units [g*16, g*16+16). Both the 64KB Whh slice and
// the full 64KB h matrix live in LDS (148KB/WG -- CDNA5 320KB WGP LDS), so the
// inner K loop (32 WMMAs) runs with no barriers at all. h is refreshed once
// per timestep with 16 async-b128 per thread.
__global__ __launch_bounds__(256)
void lstm_scan_k(const float* __restrict__ xg,    // [BT, 2048], row = b*T+t
                 const u16* __restrict__ Whh,     // [2048, 512] bf16
                 const int* __restrict__ lens,
                 u16* __restrict__ hbuf,          // [64, 512] bf16 state
                 u16* __restrict__ out,           // [BT, 1024] bf16
                 int colofs, int reversed,
                 int* barC, int* barG) {
    extern __shared__ __align__(16) char smem[];
    u16*   WsL  = (u16*)smem;                               // 64x512 = 65536 B
    u16*   Hs   = (u16*)(smem + 65536);                     // 64x512 = 65536 B
    float* gS   = (float*)(smem + 131072);                  // 4x64x16 = 16384 B
    float* cS   = (float*)(smem + 131072 + 16384);          // 64x16   =  4096 B
    int*   lenS = (int*)(smem + 131072 + 16384 + 4096);     // 64 ints =   256 B
    int tid = threadIdx.x, lane = tid & 31, wave = tid >> 5;
    int jbase = blockIdx.x * 16;

    // preload Whh slice: LDS row (gate*16+jl) <- Whh row (gate*512 + jbase + jl)
    for (int i = tid; i < (64 * 512) / 8; i += 256) {
        int r = i >> 6, cg = (i & 63) * 8;
        int gate = r >> 4, jl = r & 15;
        *(uint4*)&WsL[r * 512 + cg] =
            *(const uint4*)&Whh[(size_t)(gate * 512 + jbase + jl) * 512 + cg];
    }
    if (tid < 64) lenS[tid] = lens[tid];
    for (int i = tid; i < 64 * 16; i += 256) {
        cS[i] = 0.0f;
        int b = i >> 4, jl = i & 15;
        hbuf[b * 512 + jbase + jl] = (u16)0;
    }
    __threadfence();
    grid_barrier(barC, barG, gridDim.x, tid);

    int gate = wave & 3, mp = wave >> 2;   // acc0: rows [mp*32, +16), acc1: +16
    for (int t = 0; t < Tv; ++t) {
        // stage full 64x512 h into LDS (async, 16 x b128 per thread)
#pragma unroll
        for (int q = 0; q < 16; ++q) {
            int chunk = q * 256 + tid;           // 0..4095 chunks of 8 bf16
            int r = chunk >> 6, cg = (chunk & 63) * 8;
            async_b128(lds_off(&Hs[r * 512 + cg]),
                       (unsigned long long)(uintptr_t)&hbuf[r * 512 + cg]);
        }
        wait_async0();
        __syncthreads();

        v8f acc0 = {}, acc1 = {};
        int khalf = lane >> 4, l16 = lane & 15;
        int wr = gate * 16 + l16;
#pragma unroll
        for (int ki = 0; ki < Hv / 32; ++ki) {    // no barriers inside
            int kk = ki * 32;
            v16bf bfr = frag_ld(&WsL[wr * 512 + kk + khalf * 8],
                                &WsL[wr * 512 + kk + 16 + khalf * 8]);
            v16bf a0 = frag_ld(&Hs[(mp * 32 + l16) * 512 + kk + khalf * 8],
                               &Hs[(mp * 32 + l16) * 512 + kk + 16 + khalf * 8]);
            v16bf a1 = frag_ld(&Hs[(mp * 32 + 16 + l16) * 512 + kk + khalf * 8],
                               &Hs[(mp * 32 + 16 + l16) * 512 + kk + 16 + khalf * 8]);
            acc0 = wmma_bf16(a0, bfr, acc0);
            acc1 = wmma_bf16(a1, bfr, acc1);
        }

        // add precomputed xg (includes both biases) and park gates in LDS
        int col = gate * 512 + jbase + l16;
#pragma unroll
        for (int r = 0; r < 8; ++r) {
            int b0 = mp * 32 + r + 8 * khalf;
            int b1 = b0 + 16;
            int len0 = lenS[b0]; int ts0 = reversed ? ((t < len0) ? (len0 - 1 - t) : t) : t;
            int len1 = lenS[b1]; int ts1 = reversed ? ((t < len1) ? (len1 - 1 - t) : t) : t;
            gS[(gate * 64 + b0) * 16 + l16] = acc0[r] + xg[(size_t)(b0 * Tv + ts0) * G4H + col];
            gS[(gate * 64 + b1) * 16 + l16] = acc1[r] + xg[(size_t)(b1 * Tv + ts1) * G4H + col];
            if (r == 0 && t + 1 < Tv) {   // prefetch next step's gate rows
                int tn = t + 1;
                int tp0 = reversed ? ((tn < len0) ? (len0 - 1 - tn) : tn) : tn;
                __builtin_prefetch(&xg[(size_t)(b0 * Tv + tp0) * G4H + col], 0, 0);
            }
        }
        __syncthreads();
        // cell update for this WG's 16 hidden units x 64 batch rows
        for (int i = tid; i < 1024; i += 256) {
            int b = i >> 4, jl = i & 15;
            float ig = gS[(0 * 64 + b) * 16 + jl];
            float fg = gS[(1 * 64 + b) * 16 + jl];
            float gg = gS[(2 * 64 + b) * 16 + jl];
            float og = gS[(3 * 64 + b) * 16 + jl];
            float c = sigf(fg) * cS[i] + sigf(ig) * tanhf(gg);
            cS[i] = c;
            float h = sigf(og) * tanhf(c);
            u16 hb = f2bf(h);
            hbuf[b * 512 + jbase + jl] = hb;
            int len = lenS[b]; int ts = reversed ? ((t < len) ? (len - 1 - t) : t) : t;
            out[(size_t)(b * Tv + ts) * 1024 + colofs + jbase + jl] = hb;
        }
        __threadfence();
        grid_barrier(barC, barG, gridDim.x, tid);
    }
}

// ---------- emissions: [BT,21] = x2[BT,1024](bf16) @ W_tag[21,1024]^T + b_tag ----------
__global__ __launch_bounds__(256)
void emissions_k(const u16* __restrict__ x2, const float* __restrict__ Wt,
                 const float* __restrict__ bt, float* __restrict__ em) {
    int idx = blockIdx.x * 256 + threadIdx.x;
    if (idx >= BT * Lv) return;
    int row = idx / Lv, l = idx - row * Lv;
    const u16* xr = x2 + (size_t)row * 1024;
    const float* wr = Wt + (size_t)l * 1024;
    float s = bt[l];
#pragma unroll 8
    for (int k = 0; k < 1024; ++k) s += bf2f(xr[k]) * wr[k];
    em[idx] = s;
}

// ---------- CRF gold-path score ----------
__global__ void crf_score_k(const int* __restrict__ labels, const int* __restrict__ lens,
                            const float* __restrict__ em, const float* __restrict__ startT,
                            const float* __restrict__ endT, const float* __restrict__ trans,
                            float* out) {
    int b = threadIdx.x;
    if (b >= Bv) return;
    int len = lens[b];
    int prev = labels[b * Tv];
    float s = startT[prev] + em[(size_t)(b * Tv) * Lv + prev];
    for (int t = 1; t < Tv; ++t) {
        int lab = labels[b * Tv + t];
        if (t < len) s += em[(size_t)(b * Tv + t) * Lv + lab] + trans[prev * Lv + lab];
        prev = lab;
    }
    s += endT[labels[b * Tv + (len - 1)]];
    atomicAdd(out, -s / (float)Bv);
}

// ---------- CRF forward algorithm (logZ), one wave per batch row ----------
__global__ __launch_bounds__(32)
void crf_alpha_k(const int* __restrict__ lens, const float* __restrict__ em,
                 const float* __restrict__ startT, const float* __restrict__ endT,
                 const float* __restrict__ trans, float* out) {
    __shared__ float trS[Lv * Lv];
    __shared__ float aS[Lv];
    int b = blockIdx.x, j = threadIdx.x;
    for (int i = j; i < Lv * Lv; i += 32) trS[i] = trans[i];
    int len = lens[b];
    float alpha = (j < Lv) ? (startT[j] + em[(size_t)(b * Tv) * Lv + j]) : -3.0e38f;
    for (int t = 1; t < Tv; ++t) {
        __syncthreads();
        if (j < Lv) aS[j] = alpha;
        __syncthreads();
        if (j < Lv && t < len) {
            float m = -3.0e38f;
#pragma unroll
            for (int i = 0; i < Lv; ++i) m = fmaxf(m, aS[i] + trS[i * Lv + j]);
            float ssum = 0.0f;
#pragma unroll
            for (int i = 0; i < Lv; ++i) ssum += __expf(aS[i] + trS[i * Lv + j] - m);
            alpha = m + __logf(ssum) + em[(size_t)(b * Tv + t) * Lv + j];
        }
    }
    __syncthreads();
    if (j < Lv) aS[j] = alpha + endT[j];
    __syncthreads();
    if (j == 0) {
        float m = -3.0e38f;
        for (int i = 0; i < Lv; ++i) m = fmaxf(m, aS[i]);
        float ssum = 0.0f;
        for (int i = 0; i < Lv; ++i) ssum += __expf(aS[i] - m);
        atomicAdd(out, (m + __logf(ssum)) / (float)Bv);
    }
}

// ---------- host orchestration ----------
extern "C" void kernel_launch(void* const* d_in, const int* in_sizes, int n_in,
                              void* d_out, int out_size, void* d_ws, size_t ws_size,
                              hipStream_t stream) {
    (void)in_sizes; (void)n_in; (void)out_size; (void)ws_size;
    const int*   input_ids = (const int*)d_in[0];
    const int*   seq_lens  = (const int*)d_in[1];
    const int*   labels    = (const int*)d_in[2];
    const float* embedding = (const float*)d_in[3];
    // lstm_params flattened: ((fw,bw) layer1, (fw,bw) layer2), each (Wih,Whh,bih,bhh)
    const float* Wih[4]; const float* Whh[4]; const float* bih[4]; const float* bhh[4];
    for (int p = 0; p < 4; ++p) {
        Wih[p] = (const float*)d_in[4 + 4 * p + 0];
        Whh[p] = (const float*)d_in[4 + 4 * p + 1];
        bih[p] = (const float*)d_in[4 + 4 * p + 2];
        bhh[p] = (const float*)d_in[4 + 4 * p + 3];
    }
    const float* W_tag  = (const float*)d_in[20];
    const float* b_tag  = (const float*)d_in[21];
    const float* startT = (const float*)d_in[22];
    const float* endT   = (const float*)d_in[23];
    const float* trans  = (const float*)d_in[24];
    float* out = (float*)d_out;

    // workspace carve-out
    char* ws = (char*)d_ws; size_t off = 0;
    auto carve = [&](size_t bytes) -> void* {
        void* p = ws + off; off += (bytes + 255) & ~(size_t)255; return p;
    };
    u16*   x0    = (u16*)  carve((size_t)BT * EP * 2);
    float* xg    = (float*)carve((size_t)BT * G4H * 4);
    u16*   x1    = (u16*)  carve((size_t)BT * 1024 * 2);
    u16*   x2    = (u16*)  carve((size_t)BT * 1024 * 2);
    float* em    = (float*)carve((size_t)BT * Lv * 4);
    u16*   hbuf  = (u16*)  carve((size_t)Bv * Hv * 2);
    u16*   wih_b[4]; u16* whh_b[4];
    int    wihK[4]  = { EP, EP, 1024, 1024 };
    int    wihKs[4] = { Ev, Ev, 1024, 1024 };
    for (int p = 0; p < 4; ++p) {
        wih_b[p] = (u16*)carve((size_t)G4H * wihK[p] * 2);
        whh_b[p] = (u16*)carve((size_t)G4H * Hv * 2);
    }
    int* bar = (int*)carve(256);

    zero_init_k<<<1, 64, 0, stream>>>(out, bar);

    // weight conversion (f32 -> bf16, Wih layer1 padded 300->320 with zeros)
    for (int p = 0; p < 4; ++p) {
        cvt_pad_k<<<(G4H * wihK[p] + 255) / 256, 256, 0, stream>>>(Wih[p], wih_b[p], G4H, wihKs[p], wihK[p]);
        cvt_pad_k<<<(G4H * Hv + 255) / 256, 256, 0, stream>>>(Whh[p], whh_b[p], G4H, Hv, Hv);
    }

    // embedding gather -> x0 bf16 [BT, 320]
    embed_gather_k<<<(BT * EP + 255) / 256, 256, 0, stream>>>(input_ids, embedding, x0);

    const size_t gemmShm = 2 * (GM * GK + GN * GK) * sizeof(u16);           // 64 KB
    const size_t scanShm = 65536 + 65536 + 16384 + 4096 + 256;              // ~148 KB
    dim3 ggrid(G4H / GN, BT / GM);   // (16, 128)

    // ---- layer 1 forward / backward ----
    wmma_gemm_bias_k<<<ggrid, 256, gemmShm, stream>>>(x0, EP, wih_b[0], EP, bih[0], bhh[0], xg, BT, G4H, EP);
    lstm_scan_k<<<Hv / 16, 256, scanShm, stream>>>(xg, whh_b[0], seq_lens, hbuf, x1, 0, 0, bar, bar + 1);
    wmma_gemm_bias_k<<<ggrid, 256, gemmShm, stream>>>(x0, EP, wih_b[1], EP, bih[1], bhh[1], xg, BT, G4H, EP);
    lstm_scan_k<<<Hv / 16, 256, scanShm, stream>>>(xg, whh_b[1], seq_lens, hbuf, x1, Hv, 1, bar, bar + 1);

    // ---- layer 2 forward / backward (input = x1 [BT, 1024]) ----
    wmma_gemm_bias_k<<<ggrid, 256, gemmShm, stream>>>(x1, 1024, wih_b[2], 1024, bih[2], bhh[2], xg, BT, G4H, 1024);
    lstm_scan_k<<<Hv / 16, 256, scanShm, stream>>>(xg, whh_b[2], seq_lens, hbuf, x2, 0, 0, bar, bar + 1);
    wmma_gemm_bias_k<<<ggrid, 256, gemmShm, stream>>>(x1, 1024, wih_b[3], 1024, bih[3], bhh[3], xg, BT, G4H, 1024);
    lstm_scan_k<<<Hv / 16, 256, scanShm, stream>>>(xg, whh_b[3], seq_lens, hbuf, x2, Hv, 1, bar, bar + 1);

    // ---- emissions + CRF ----
    emissions_k<<<(BT * Lv + 255) / 256, 256, 0, stream>>>(x2, W_tag, b_tag, em);
    crf_score_k<<<1, 64, 0, stream>>>(labels, seq_lens, em, startT, endT, trans, out);
    crf_alpha_k<<<Bv, 32, 0, stream>>>(seq_lens, em, startT, endT, trans, out);
}
